// MultiHeadAttention_29300266893958
// MI455X (gfx1250) — compile-verified, matrix operand order checked by reading, not run on hardware
//
#include <hip/hip_runtime.h>
#include <hip/hip_bf16.h>

typedef __bf16 bf16;
typedef __attribute__((ext_vector_type(16))) __bf16 bf16x16;
typedef __attribute__((ext_vector_type(8)))  __bf16 bf16x8;
typedef __attribute__((ext_vector_type(8)))  float  v8f;
typedef int v4i __attribute__((vector_size(4 * sizeof(int))));

#define D_MODEL 768
#define N_SEQ   4096
#define BATCH   2
#define HEADS   12
#define HDIM    64
#define M_TOTAL (BATCH * N_SEQ)   // 8192 total rows

// Async global->LDS copy (CDNA5 GLOBAL_LOAD_ASYNC_TO_LDS_B128, ASYNCcnt) with
// a safe fallback if this toolchain doesn't expose the builtins.
#if defined(__HIP_DEVICE_COMPILE__) && \
    __has_builtin(__builtin_amdgcn_global_load_async_to_lds_b128) && \
    __has_builtin(__builtin_amdgcn_s_wait_asynccnt)
#define HAVE_ASYNC 1
#else
#define HAVE_ASYNC 0
#endif

__device__ __forceinline__ void async_copy_b128(const bf16* g, bf16* l) {
#if HAVE_ASYNC
  __builtin_amdgcn_global_load_async_to_lds_b128(
      (__attribute__((address_space(1))) v4i*)g,
      (__attribute__((address_space(3))) v4i*)l, 0, 0);
#else
  *(bf16x8*)l = *(const bf16x8*)g;
#endif
}

__device__ __forceinline__ void async_wait() {
#if HAVE_ASYNC
  __builtin_amdgcn_s_wait_asynccnt(0);
#endif
}

// ---------------------------------------------------------------------------
// Fragment loader: 16x32 bf16 A-fragment (or 32x16 B-fragment by symmetry)
// from K-contiguous storage. Per CDNA5 ISA 7.12.2 (16-bit A 16x32):
//   lanes 0-15 : row = lane, elements K = {0..7, 16..23}
//   lanes 16-31: row = lane-16, elements K = {8..15, 24..31}
// stride in bf16 elements; rows must be 16B aligned (stride*2 % 16 == 0).
// ---------------------------------------------------------------------------
__device__ __forceinline__ bf16x16 load_frag(const bf16* base, int stride, int lane) {
  const int r  = lane & 15;
  const int c0 = (lane < 16) ? 0 : 8;
  const bf16* p = base + r * stride + c0;
  bf16x8 lo = *(const bf16x8*)(p);
  bf16x8 hi = *(const bf16x8*)(p + 16);
  bf16x16 f;
#pragma unroll
  for (int i = 0; i < 8; ++i) { f[i] = lo[i]; f[i + 8] = hi[i]; }
  return f;
}

__device__ __forceinline__ v8f wmma_bf16(bf16x16 a, bf16x16 b, v8f c) {
  return __builtin_amdgcn_wmma_f32_16x16x32_bf16(false, a, false, b, (short)0, c,
                                                 false, false);
}

// ---------------------------------------------------------------------------
// Kernel 0: convert 4 weight matrices [k][n] f32 -> transposed bf16 [n][k]
// ---------------------------------------------------------------------------
__global__ __launch_bounds__(256) void prep_weights_kernel(
    const float* __restrict__ wq, const float* __restrict__ wk,
    const float* __restrict__ wv, const float* __restrict__ wo,
    bf16* __restrict__ dst) {
  const int z = blockIdx.y;
  const float* src = (z == 0) ? wq : (z == 1) ? wk : (z == 2) ? wv : wo;
  bf16* d = dst + (size_t)z * D_MODEL * D_MODEL;
  int i = blockIdx.x * 256 + threadIdx.x;
  if (i < D_MODEL * D_MODEL) {
    int n = i / D_MODEL, k = i - n * D_MODEL;
    d[i] = (bf16)src[(size_t)k * D_MODEL + n];  // d[n][k] = w[k][n]
  }
}

// ---------------------------------------------------------------------------
// Kernel 1: QKV projection. out[z] = x @ W[z], stored as [b][h][n][d] bf16.
// Block: 256 thr (8 waves), tile M=128 x N=128, K step 32.
// Wave (wr, wc) computes a 32x64 subtile = 8 WMMA C tiles.
// ---------------------------------------------------------------------------
__global__ __launch_bounds__(256) void qkv_proj_kernel(
    const float* __restrict__ x, const bf16* __restrict__ wt_base,
    bf16* __restrict__ out_base) {
  const int z = blockIdx.z;
  const bf16* Wt = wt_base + (size_t)z * D_MODEL * D_MODEL;    // [n][k]
  bf16* out = out_base + (size_t)z * BATCH * HEADS * N_SEQ * HDIM;

  const int m0 = blockIdx.x * 128;
  const int n0 = blockIdx.y * 128;
  const int tid = threadIdx.x;
  const int wave = tid >> 5, lane = tid & 31;
  const int wr = wave & 3, wc = wave >> 2;   // 4x2 wave grid

  __shared__ __align__(16) bf16 lds_a[128][40];  // 80B rows (16B aligned)
  __shared__ __align__(16) bf16 lds_b[128][40];

  v8f acc[2][4] = {};

  for (int k0 = 0; k0 < D_MODEL; k0 += 32) {
    // A: f32 -> bf16 conversion in flight (cannot be async-copied)
#pragma unroll
    for (int it = 0; it < 16; ++it) {            // 128*32 / 256
      int i = tid + it * 256;
      int r = i >> 5, c = i & 31;
      lds_a[r][c] = (bf16)x[(size_t)(m0 + r) * D_MODEL + k0 + c];
    }
    // B: straight bf16 copy -> async to LDS
#pragma unroll
    for (int it = 0; it < 2; ++it) {             // 512 16B chunks / 256
      int ch = tid + it * 256;
      int r = ch >> 2, c8 = (ch & 3) * 8;
      async_copy_b128(Wt + (size_t)(n0 + r) * D_MODEL + k0 + c8, &lds_b[r][c8]);
    }
    async_wait();
    __syncthreads();
    bf16x16 a0 = load_frag(&lds_a[wr * 32][0],      40, lane);
    bf16x16 a1 = load_frag(&lds_a[wr * 32 + 16][0], 40, lane);
#pragma unroll
    for (int tj = 0; tj < 4; ++tj) {
      bf16x16 b = load_frag(&lds_b[wc * 64 + tj * 16][0], 40, lane);
      acc[0][tj] = wmma_bf16(a0, b, acc[0][tj]);
      acc[1][tj] = wmma_bf16(a1, b, acc[1][tj]);
    }
    __syncthreads();
  }

  // C layout: VGPR i holds row i (lanes 0-15) / row i+8 (lanes 16-31), col = lane%16
#pragma unroll
  for (int ti = 0; ti < 2; ++ti)
#pragma unroll
    for (int tj = 0; tj < 4; ++tj)
#pragma unroll
      for (int i = 0; i < 8; ++i) {
        int row = m0 + wr * 32 + ti * 16 + ((lane < 16) ? i : i + 8);
        int col = n0 + wc * 64 + tj * 16 + (lane & 15);
        int b = row >> 12, n = row & (N_SEQ - 1);
        int h = col >> 6,  d = col & 63;
        size_t idx = (((size_t)b * HEADS + h) * N_SEQ + n) * HDIM + d;
        out[idx] = (bf16)acc[ti][tj][i];
      }
}

// ---------------------------------------------------------------------------
// Kernel 2: causal flash attention over one (b,h). Block = 128 query rows,
// 8 waves x 16 rows each. Key tiles of 64, cooperatively staged in LDS.
// Per key tile per wave: 8 WMMA for S = Q K^T, 8 WMMA for P V.
// ---------------------------------------------------------------------------
__global__ __launch_bounds__(256) void attn_kernel(
    const bf16* __restrict__ qkv, bf16* __restrict__ ctx) {
  const int bh = blockIdx.y;                                   // b*HEADS + h
  const size_t QS = (size_t)BATCH * HEADS * N_SEQ * HDIM;
  const bf16* Q  = qkv +           (size_t)bh * N_SEQ * HDIM;
  const bf16* Kp = qkv + QS +      (size_t)bh * N_SEQ * HDIM;
  const bf16* Vp = qkv + 2 * QS +  (size_t)bh * N_SEQ * HDIM;

  const int q0b = blockIdx.x * 128;
  const int tid = threadIdx.x;
  const int wave = tid >> 5, lane = tid & 31;
  const int q0w = q0b + wave * 16;

  __shared__ __align__(16) bf16 lds_k[64][72];      // [key][d], 144B rows
  __shared__ __align__(16) bf16 lds_vt[64][72];     // [d][key]
  __shared__ __align__(16) bf16 lds_p[8][16][72];   // per-wave P buffer

  // Q kept in registers: two 16x32 A-fragments (d 0..31, 32..63)
  bf16x16 qa0 = load_frag(Q + (size_t)q0w * HDIM,      HDIM, lane);
  bf16x16 qa1 = load_frag(Q + (size_t)q0w * HDIM + 32, HDIM, lane);

  float m_i[8], l_i[8];
  v8f o[4] = {};
#pragma unroll
  for (int i = 0; i < 8; ++i) { m_i[i] = -1e30f; l_i[i] = 0.0f; }
  const float sc = 0.125f * 1.44269504088896f;      // 1/sqrt(64) * log2(e)
  const int my_last = q0w + 15;
  const int kend = q0b + 128;                       // causal bound for block

  for (int kt = 0; kt < kend; kt += 64) {
    // K tile [key][d]: straight bf16 copy -> async to LDS
#pragma unroll
    for (int it = 0; it < 2; ++it) {                // 512 16B chunks / 256
      int ch = tid + it * 256;
      int r = ch >> 3, c8 = (ch & 7) * 8;
      async_copy_b128(Kp + (size_t)(kt + r) * HDIM + c8, &lds_k[r][c8]);
    }
    // V^T tile [d][key]: transposed in flight (manual)
#pragma unroll
    for (int it = 0; it < 16; ++it) {               // 64*64 / 256
      int i = tid + it * 256;
      int r = i >> 6, c = i & 63;                   // r=key-in-tile, c=d
      lds_vt[c][r] = Vp[(size_t)(kt + r) * HDIM + c];
    }
    async_wait();
    __syncthreads();

    if (kt <= my_last) {
      // S = Q K^T : four 16x16 C tiles (key groups of 16)
      v8f s[4];
#pragma unroll
      for (int g = 0; g < 4; ++g) {
        s[g] = (v8f){};
        bf16x16 kbl = load_frag(&lds_k[g * 16][0],  72, lane);
        bf16x16 kbh = load_frag(&lds_k[g * 16][32], 72, lane);
        s[g] = wmma_bf16(qa0, kbl, s[g]);
        s[g] = wmma_bf16(qa1, kbh, s[g]);
      }

      // causal mask + scale (exp2 domain), then online softmax
      float p[4][8], mx[8];
#pragma unroll
      for (int i = 0; i < 8; ++i) {
        int row = q0w + ((lane < 16) ? i : i + 8);
        float m = m_i[i];
#pragma unroll
        for (int g = 0; g < 4; ++g) {
          int col = kt + g * 16 + (lane & 15);
          float t = (col <= row) ? s[g][i] * sc : -1e30f;
          p[g][i] = t;
          m = fmaxf(m, t);
        }
        mx[i] = m;
      }
#pragma unroll
      for (int off = 1; off < 16; off <<= 1)
#pragma unroll
        for (int i = 0; i < 8; ++i)
          mx[i] = fmaxf(mx[i], __shfl_xor(mx[i], off));

      float rs[8];
#pragma unroll
      for (int i = 0; i < 8; ++i) {
        float r = 0.0f;
#pragma unroll
        for (int g = 0; g < 4; ++g) {
          float e = exp2f(p[g][i] - mx[i]);
          p[g][i] = e;
          r += e;
        }
        rs[i] = r;
      }
#pragma unroll
      for (int off = 1; off < 16; off <<= 1)
#pragma unroll
        for (int i = 0; i < 8; ++i)
          rs[i] += __shfl_xor(rs[i], off);

#pragma unroll
      for (int i = 0; i < 8; ++i) {
        float corr = exp2f(m_i[i] - mx[i]);
        l_i[i] = l_i[i] * corr + rs[i];
        m_i[i] = mx[i];
#pragma unroll
        for (int t = 0; t < 4; ++t) o[t][i] *= corr;
      }

      // C-layout P -> A-layout via per-wave LDS buffer (wave-private, in-order)
      bf16 (*P)[72] = lds_p[wave];
#pragma unroll
      for (int i = 0; i < 8; ++i) {
        int r = (lane < 16) ? i : i + 8;
#pragma unroll
        for (int g = 0; g < 4; ++g)
          P[r][g * 16 + (lane & 15)] = (bf16)p[g][i];
      }
      bf16x16 pa0 = load_frag(&P[0][0],  72, lane);   // keys 0..31
      bf16x16 pa1 = load_frag(&P[0][32], 72, lane);   // keys 32..63
#pragma unroll
      for (int t = 0; t < 4; ++t) {
        bf16x16 vbl = load_frag(&lds_vt[t * 16][0],  72, lane);
        bf16x16 vbh = load_frag(&lds_vt[t * 16][32], 72, lane);
        o[t] = wmma_bf16(pa0, vbl, o[t]);
        o[t] = wmma_bf16(pa1, vbh, o[t]);
      }
    }
    __syncthreads();
  }

  // normalize and store ctx as [b][n][h*64+d] bf16
  const int b = bh / HEADS, h = bh % HEADS;
#pragma unroll
  for (int i = 0; i < 8; ++i) {
    int n = q0w + ((lane < 16) ? i : i + 8);
    float inv = 1.0f / l_i[i];
#pragma unroll
    for (int t = 0; t < 4; ++t) {
      int d = t * 16 + (lane & 15);
      size_t idx = ((size_t)b * N_SEQ + n) * D_MODEL + h * HDIM + d;
      ctx[idx] = (bf16)(o[t][i] * inv);
    }
  }
}

// ---------------------------------------------------------------------------
// Kernel 3: out = ctx @ w_out + b_out (fp32 output). 128x128 tile.
// ---------------------------------------------------------------------------
__global__ __launch_bounds__(256) void out_proj_kernel(
    const bf16* __restrict__ ctx, const bf16* __restrict__ wo_t,
    const float* __restrict__ bias, float* __restrict__ out) {
  const int m0 = blockIdx.x * 128;
  const int n0 = blockIdx.y * 128;
  const int tid = threadIdx.x;
  const int wave = tid >> 5, lane = tid & 31;
  const int wr = wave & 3, wc = wave >> 2;

  __shared__ __align__(16) bf16 lds_a[128][40];
  __shared__ __align__(16) bf16 lds_b[128][40];

  v8f acc[2][4] = {};

  for (int k0 = 0; k0 < D_MODEL; k0 += 32) {
    // both operands are straight bf16 copies -> async to LDS
#pragma unroll
    for (int it = 0; it < 2; ++it) {
      int ch = tid + it * 256;
      int r = ch >> 2, c8 = (ch & 3) * 8;
      async_copy_b128(ctx  + (size_t)(m0 + r) * D_MODEL + k0 + c8, &lds_a[r][c8]);
      async_copy_b128(wo_t + (size_t)(n0 + r) * D_MODEL + k0 + c8, &lds_b[r][c8]);
    }
    async_wait();
    __syncthreads();
    bf16x16 a0 = load_frag(&lds_a[wr * 32][0],      40, lane);
    bf16x16 a1 = load_frag(&lds_a[wr * 32 + 16][0], 40, lane);
#pragma unroll
    for (int tj = 0; tj < 4; ++tj) {
      bf16x16 b = load_frag(&lds_b[wc * 64 + tj * 16][0], 40, lane);
      acc[0][tj] = wmma_bf16(a0, b, acc[0][tj]);
      acc[1][tj] = wmma_bf16(a1, b, acc[1][tj]);
    }
    __syncthreads();
  }

#pragma unroll
  for (int ti = 0; ti < 2; ++ti)
#pragma unroll
    for (int tj = 0; tj < 4; ++tj)
#pragma unroll
      for (int i = 0; i < 8; ++i) {
        int row = m0 + wr * 32 + ti * 16 + ((lane < 16) ? i : i + 8);
        int col = n0 + wc * 64 + tj * 16 + (lane & 15);
        out[(size_t)row * D_MODEL + col] = acc[ti][tj][i] + bias[col];
      }
}

// ---------------------------------------------------------------------------
extern "C" void kernel_launch(void* const* d_in, const int* in_sizes, int n_in,
                              void* d_out, int out_size, void* d_ws, size_t ws_size,
                              hipStream_t stream) {
  const float* x  = (const float*)d_in[0];
  const float* wq = (const float*)d_in[1];
  const float* wk = (const float*)d_in[2];
  const float* wv = (const float*)d_in[3];
  const float* wo = (const float*)d_in[4];
  const float* bo = (const float*)d_in[5];
  float* out = (float*)d_out;

  bf16* ws = (bf16*)d_ws;
  const size_t W  = (size_t)D_MODEL * D_MODEL;              // 589,824 per weight
  const size_t QS = (size_t)BATCH * HEADS * N_SEQ * HDIM;   // 6,291,456 per q/k/v
  bf16* wt  = ws;                  // 4 weight matrices, transposed bf16
  bf16* qkv = ws + 4 * W;          // q, k, v in [b][h][n][d]
  bf16* ctx = ws + 4 * W + 3 * QS; // [b][n][768]
  if (ws_size < (4 * W + 4 * QS) * sizeof(bf16)) return;   // ~55 MB needed

  prep_weights_kernel<<<dim3((unsigned)((W + 255) / 256), 4), 256, 0, stream>>>(
      wq, wk, wv, wo, wt);
  qkv_proj_kernel<<<dim3(M_TOTAL / 128, D_MODEL / 128, 3), 256, 0, stream>>>(
      x, wt, qkv);
  attn_kernel<<<dim3(N_SEQ / 128, BATCH * HEADS), 256, 0, stream>>>(qkv, ctx);
  out_proj_kernel<<<dim3(M_TOTAL / 128, D_MODEL / 128), 256, 0, stream>>>(
      ctx, wt + 3 * W, bo, out);
}
